// QLinear_42949672960247
// MI455X (gfx1250) — 8-bit matrix operand order repaired and checked by reading, compiled, not run on hardware
//
#include <hip/hip_runtime.h>

typedef __attribute__((ext_vector_type(8))) int v8i;

#define M_DIM 8192
#define N_DIM 4096
#define K_DIM 4096

__device__ __forceinline__ int pack4_rne(float a, float b, float c, float d, float inv) {
    int ia = (int)__builtin_rintf(a * inv);
    int ib = (int)__builtin_rintf(b * inv);
    int ic = (int)__builtin_rintf(c * inv);
    int id = (int)__builtin_rintf(d * inv);
    return (ia & 255) | ((ib & 255) << 8) | ((ic & 255) << 16) | ((id & 255) << 24);
}

__global__ __launch_bounds__(256) void quant_rows_kernel(
    const float* __restrict__ x, signed char* __restrict__ xq,
    float* __restrict__ sx)
{
    __shared__ float red[256];
    const int row = blockIdx.x;
    const int t   = threadIdx.x;

    const float4* xr = (const float4*)(x + (size_t)row * K_DIM);
    float4 v0 = xr[t * 4 + 0];
    float4 v1 = xr[t * 4 + 1];
    float4 v2 = xr[t * 4 + 2];
    float4 v3 = xr[t * 4 + 3];

    float am = 0.0f;
    am = fmaxf(am, fmaxf(fmaxf(fabsf(v0.x), fabsf(v0.y)), fmaxf(fabsf(v0.z), fabsf(v0.w))));
    am = fmaxf(am, fmaxf(fmaxf(fabsf(v1.x), fabsf(v1.y)), fmaxf(fabsf(v1.z), fabsf(v1.w))));
    am = fmaxf(am, fmaxf(fmaxf(fabsf(v2.x), fabsf(v2.y)), fmaxf(fabsf(v2.z), fabsf(v2.w))));
    am = fmaxf(am, fmaxf(fmaxf(fabsf(v3.x), fabsf(v3.y)), fmaxf(fabsf(v3.z), fabsf(v3.w))));

    red[t] = am;
    __syncthreads();
#pragma unroll
    for (int s = 128; s > 0; s >>= 1) {
        if (t < s) red[t] = fmaxf(red[t], red[t + s]);
        __syncthreads();
    }
    am = red[0];

    const float inv = (am > 0.0f) ? (127.0f / am) : 0.0f;
    if (t == 0) sx[row] = am * (1.0f / 127.0f);

    int4 o;
    o.x = pack4_rne(v0.x, v0.y, v0.z, v0.w, inv);
    o.y = pack4_rne(v1.x, v1.y, v1.z, v1.w, inv);
    o.z = pack4_rne(v2.x, v2.y, v2.z, v2.w, inv);
    o.w = pack4_rne(v3.x, v3.y, v3.z, v3.w, inv);
    ((int4*)(xq + (size_t)row * K_DIM))[t] = o;
}

__global__ __launch_bounds__(256) void int8_wmma_gemm_kernel(
    const signed char* __restrict__ xq, const signed char* __restrict__ cb,
    const float* __restrict__ sx, const float* __restrict__ scb,
    float* __restrict__ y)
{
    const int lane  = threadIdx.x & 31;
    const int wave  = threadIdx.x >> 5;
    const int l15   = lane & 15;
    const int khalf = lane >> 4;

    const int rowBase = blockIdx.y * 128 + (wave >> 2) * 64;
    const int colBase = blockIdx.x * 256 + (wave & 3) * 64;

    v8i acc[4][4];
#pragma unroll
    for (int mi = 0; mi < 4; ++mi)
#pragma unroll
        for (int ni = 0; ni < 4; ++ni) {
            v8i z = {0, 0, 0, 0, 0, 0, 0, 0};
            acc[mi][ni] = z;
        }

    const signed char* aBase[4];
    const signed char* bBase[4];
#pragma unroll
    for (int mi = 0; mi < 4; ++mi)
        aBase[mi] = xq + (size_t)(rowBase + mi * 16 + l15) * K_DIM + khalf * 16;
#pragma unroll
    for (int ni = 0; ni < 4; ++ni)
        bBase[ni] = cb + (size_t)(colBase + ni * 16 + l15) * K_DIM + khalf * 16;

    for (int kb = 0; kb < K_DIM; kb += 64) {
        v8i a[4], b[4];
#pragma unroll
        for (int mi = 0; mi < 4; ++mi) {
            const int4* p = (const int4*)(aBase[mi] + kb);
            int4 c0 = p[0], c1 = p[2];
            a[mi][0] = c0.x; a[mi][1] = c0.y; a[mi][2] = c0.z; a[mi][3] = c0.w;
            a[mi][4] = c1.x; a[mi][5] = c1.y; a[mi][6] = c1.z; a[mi][7] = c1.w;
        }
#pragma unroll
        for (int ni = 0; ni < 4; ++ni) {
            const int4* p = (const int4*)(bBase[ni] + kb);
            int4 c0 = p[0], c1 = p[2];
            b[ni][0] = c0.x; b[ni][1] = c0.y; b[ni][2] = c0.z; b[ni][3] = c0.w;
            b[ni][4] = c1.x; b[ni][5] = c1.y; b[ni][6] = c1.z; b[ni][7] = c1.w;
        }

        if (kb + 64 < K_DIM) {
            __builtin_prefetch(aBase[0] + kb + 64, 0, 1);
            __builtin_prefetch(bBase[0] + kb + 64, 0, 1);
        }

#pragma unroll
        for (int mi = 0; mi < 4; ++mi)
#pragma unroll
            for (int ni = 0; ni < 4; ++ni)
                acc[mi][ni] = __builtin_amdgcn_wmma_i32_16x16x64_iu8(
                     true, a[mi],  true, b[ni],
                    acc[mi][ni],  false,  false);
    }

    const float inv127 = 1.0f / 127.0f;
#pragma unroll
    for (int ni = 0; ni < 4; ++ni) {
        const int n = colBase + ni * 16 + l15;
        const float sn = scb[n] * inv127;
#pragma unroll
        for (int mi = 0; mi < 4; ++mi) {
            const int mb = rowBase + mi * 16 + khalf * 8;
#pragma unroll
            for (int r = 0; r < 8; ++r) {
                const int m = mb + r;
                y[(size_t)m * N_DIM + n] = (float)acc[mi][ni][r] * sx[m] * sn;
            }
        }
    }
}

extern "C" void kernel_launch(void* const* d_in, const int* in_sizes, int n_in,
                              void* d_out, int out_size, void* d_ws, size_t ws_size,
                              hipStream_t stream) {
    const float*       x   = (const float*)d_in[0];
    const signed char* cb  = (const signed char*)d_in[1];
    const float*       scb = (const float*)d_in[2];
    float*             y   = (float*)d_out;

    signed char* xq = (signed char*)d_ws;
    float*       sx = (float*)((char*)d_ws + (size_t)M_DIM * K_DIM);

    quant_rows_kernel<<<M_DIM, 256, 0, stream>>>(x, xq, sx);

    dim3 grid(N_DIM / 256, M_DIM / 128);
    int8_wmma_gemm_kernel<<<grid, 256, 0, stream>>>(xq, cb, sx, scb, y);
}
